// JBUStack_34711925686945
// MI455X (gfx1250) — compile-verified, matrix operand order checked by reading, not run on hardware
//
#include <hip/hip_runtime.h>
#include <hip/hip_bf16.h>

typedef __attribute__((ext_vector_type(16))) __bf16 v16bf;
typedef __attribute__((ext_vector_type(8)))  float  f32x8;
typedef unsigned int u32x4 __attribute__((ext_vector_type(4)));
typedef unsigned int u32x8 __attribute__((ext_vector_type(8)));

#define RADIUS 3
#define DIAM 7
#define KEYD 32
#define CCH 384
#define OCH 256

__device__ __forceinline__ int refl(int v, int n) {
    if (v < 0) v = -v;
    if (v >= n) v = 2 * n - 2 - v;
    return v;
}

__device__ __forceinline__ float gelu_tanh(float x) {
    float x3 = x * x * x;
    return 0.5f * x * (1.0f + tanhf(0.7978845608028654f * (x + 0.044715f * x3)));
}

__device__ __forceinline__ unsigned short f2bf(float f) {
    unsigned int u = __builtin_bit_cast(unsigned int, f);
    u += 0x7FFFu + ((u >> 16) & 1u);
    return (unsigned short)(u >> 16);
}

// Keys cubic, a = -0.5
__device__ __forceinline__ float cubw(float t) {
    t = fabsf(t);
    if (t <= 1.0f) return ((1.5f * t - 2.5f) * t) * t + 1.0f;
    if (t < 2.0f)  return ((-0.5f * t + 2.5f) * t - 4.0f) * t + 2.0f;
    return 0.0f;
}

// ---------------- adaptive average pool: guidance [B,3,768,768] -> [B,3,OH,OW]
__global__ void pool_kernel(const float* __restrict__ g, float* __restrict__ out,
                            int OH, int OW, int f, int total) {
    int i = blockIdx.x * blockDim.x + threadIdx.x;
    if (i >= total) return;
    int ox = i % OW; int t = i / OW;
    int oy = t % OH; t /= OH;
    int c = t % 3;   int b = t / 3;
    const float* src = g + (((size_t)(b * 3 + c) * 768 + (size_t)oy * f) * 768) + (size_t)ox * f;
    float s = 0.0f;
    for (int yy = 0; yy < f; ++yy) {
        const float* r = src + (size_t)yy * 768;
        for (int xx = 0; xx < f; ++xx) s += r[xx];
    }
    out[i] = s / (float)(f * f);
}

// ---------------- range projection: proj = W2 * gelu(W1*g + b1) + b2, per pixel
__global__ void rangeproj_kernel(const float* __restrict__ g,
                                 const float* __restrict__ w1, const float* __restrict__ b1,
                                 const float* __restrict__ w2, const float* __restrict__ b2,
                                 float* __restrict__ proj, int S, int total) {
    __shared__ float sw1[KEYD * 3], sb1[KEYD], sw2[KEYD * KEYD], sb2[KEYD];
    for (int i = threadIdx.x; i < KEYD * 3; i += blockDim.x) sw1[i] = w1[i];
    for (int i = threadIdx.x; i < KEYD; i += blockDim.x) { sb1[i] = b1[i]; sb2[i] = b2[i]; }
    for (int i = threadIdx.x; i < KEYD * KEYD; i += blockDim.x) sw2[i] = w2[i];
    __syncthreads();
    int i = blockIdx.x * blockDim.x + threadIdx.x;
    if (i >= total) return;
    int s = i % S; int b = i / S;
    const float* gp = g + (size_t)b * 3 * S + s;
    float g0 = gp[0], g1 = gp[S], g2 = gp[2 * (size_t)S];
    float h[KEYD];
    #pragma unroll
    for (int j = 0; j < KEYD; ++j) {
        float v = sw1[j * 3] * g0 + sw1[j * 3 + 1] * g1 + sw1[j * 3 + 2] * g2 + sb1[j];
        h[j] = gelu_tanh(v);
    }
    float* pp = proj + (size_t)b * KEYD * S + s;
    #pragma unroll
    for (int k = 0; k < KEYD; ++k) {
        float v = sb2[k];
        #pragma unroll
        for (int j = 0; j < KEYD; ++j) v += sw2[k * KEYD + j] * h[j];
        pp[(size_t)k * S] = v;
    }
}

// ---------------- per-pixel 49-way kernel: softmax(temp*logits) * gaussian, normalized
__global__ void combined_kernel(const float* __restrict__ proj,
                                const float* __restrict__ temp_p, const float* __restrict__ sigma_p,
                                float* __restrict__ comb, int GH, int GW, int total) {
    int i = blockIdx.x * blockDim.x + threadIdx.x;
    if (i >= total) return;
    int S = GH * GW;
    int s = i % S; int b = i / S;
    int y = s / GW, x = s % GW;
    const float* P = proj + (size_t)b * KEYD * S;
    float cvec[KEYD];
    #pragma unroll
    for (int k = 0; k < KEYD; ++k) cvec[k] = P[(size_t)k * S + s];
    float lg[DIAM * DIAM];
    float mx = -3.4e38f;
    for (int di = 0; di < DIAM; ++di) {
        int ry = refl(y + di - RADIUS, GH);
        for (int dj = 0; dj < DIAM; ++dj) {
            int rx = refl(x + dj - RADIUS, GW);
            const float* q = P + (size_t)ry * GW + rx;
            float d = 0.0f;
            #pragma unroll
            for (int k = 0; k < KEYD; ++k) d += cvec[k] * q[(size_t)k * S];
            lg[di * DIAM + dj] = d;
            mx = fmaxf(mx, d);
        }
    }
    float tmp = expf(temp_p[0]);
    tmp = fminf(fmaxf(tmp, 1e-4f), 1e4f);
    float sig = sigma_p[0];
    float inv2s2 = 1.0f / (2.0f * sig * sig);
    float sumE = 0.0f;
    #pragma unroll
    for (int p = 0; p < DIAM * DIAM; ++p) { lg[p] = expf(tmp * (lg[p] - mx)); sumE += lg[p]; }
    float invE = 1.0f / sumE;
    float sumC = 0.0f;
    #pragma unroll
    for (int p = 0; p < DIAM * DIAM; ++p) {
        float dy = (float)(p / DIAM - RADIUS) * (1.0f / 3.0f);
        float dx = (float)(p % DIAM - RADIUS) * (1.0f / 3.0f);
        float sp = expf(-(dx * dx + dy * dy) * inv2s2);
        lg[p] = lg[p] * invE * sp;
        sumC += lg[p];
    }
    float inv = 1.0f / fmaxf(sumC, 1e-7f);
    float* cb = comb + (size_t)b * (DIAM * DIAM) * S + s;
    #pragma unroll
    for (int p = 0; p < DIAM * DIAM; ++p) cb[(size_t)p * S] = lg[p] * inv;
}

// ---------------- bicubic 2x upsample (half-pixel, Keys a=-0.5, edge-renormalized)
__global__ void bicubic_kernel(const float* __restrict__ src, float* __restrict__ dst,
                               int H, int W, int total) {
    int i = blockIdx.x * blockDim.x + threadIdx.x;
    if (i >= total) return;
    int OW = 2 * W, OH = 2 * H;
    int ox = i % OW; int t = i / OW;
    int oy = t % OH; int bc = t / OH;
    float cy = 0.5f * (float)oy - 0.25f;
    float cx = 0.5f * (float)ox - 0.25f;
    int fy = (oy >> 1) + ((oy & 1) ? 0 : -1);
    int fx = (ox >> 1) + ((ox & 1) ? 0 : -1);
    float wy[4], wx[4]; int iy[4], ix[4];
    float sy = 0.0f, sx = 0.0f;
    #pragma unroll
    for (int k = 0; k < 4; ++k) {
        int yy = fy - 1 + k;
        float w = cubw(cy - (float)yy);
        if (yy < 0 || yy >= H) w = 0.0f;
        sy += w; wy[k] = w; iy[k] = min(max(yy, 0), H - 1);
        int xx = fx - 1 + k;
        float v = cubw(cx - (float)xx);
        if (xx < 0 || xx >= W) v = 0.0f;
        sx += v; wx[k] = v; ix[k] = min(max(xx, 0), W - 1);
    }
    const float* sp = src + (size_t)bc * H * W;
    float acc = 0.0f;
    #pragma unroll
    for (int k = 0; k < 4; ++k) {
        const float* r = sp + (size_t)iy[k] * W;
        float ra = wx[0] * r[ix[0]] + wx[1] * r[ix[1]] + wx[2] * r[ix[2]] + wx[3] * r[ix[3]];
        acc += wy[k] * ra;
    }
    dst[i] = acc / (sy * sx);
}

// ---------------- adaptive 7x7 per-pixel conv applied to all channels
__global__ void adaptive_kernel(const float* __restrict__ hr, const float* __restrict__ comb,
                                float* __restrict__ out, int C, int GH, int GW, int total) {
    int i = blockIdx.x * blockDim.x + threadIdx.x;
    if (i >= total) return;
    int S = GH * GW;
    int s = i % S; int b = i / S;
    int y = s / GW, x = s % GW;
    float k[DIAM * DIAM];
    const float* cb = comb + (size_t)b * (DIAM * DIAM) * S + s;
    #pragma unroll
    for (int p = 0; p < DIAM * DIAM; ++p) k[p] = cb[(size_t)p * S];
    int ry[DIAM], rx[DIAM];
    #pragma unroll
    for (int d = 0; d < DIAM; ++d) {
        ry[d] = refl(y + d - RADIUS, GH);
        rx[d] = refl(x + d - RADIUS, GW);
    }
    const float* hb = hr + (size_t)b * C * S;
    float* ob = out + (size_t)b * C * S + s;
    for (int c = 0; c < C; ++c) {
        const float* hc = hb + (size_t)c * S;
        float acc = 0.0f;
        #pragma unroll
        for (int d = 0; d < DIAM; ++d) {
            const float* r = hc + (size_t)ry[d] * GW;
            float a0 = 0.0f;
            #pragma unroll
            for (int e = 0; e < DIAM; ++e) a0 += k[d * DIAM + e] * r[rx[e]];
            acc += a0;
        }
        ob[(size_t)c * S] = acc;
    }
}

// ---------------- fused weight precompute: W_eff = proj_w*(0.1*fixup_w + I) (bf16), b_eff
__global__ void weff_kernel(const float* __restrict__ fw, const float* __restrict__ fb,
                            const float* __restrict__ pw, const float* __restrict__ pb,
                            unsigned short* __restrict__ Wbf, float* __restrict__ beff) {
    int i = blockIdx.x * blockDim.x + threadIdx.x;
    if (i < OCH * CCH) {
        int o = i / CCH, kk = i % CCH;
        float v = pw[(size_t)o * CCH + kk];
        for (int j = 0; j < CCH; ++j)
            v += 0.1f * pw[(size_t)o * CCH + j] * fw[(size_t)j * CCH + kk];
        Wbf[i] = f2bf(v);
    }
    if (i < OCH) {
        float v = pb[i];
        for (int j = 0; j < CCH; ++j) v += 0.1f * pw[(size_t)i * CCH + j] * fb[j];
        beff[i] = v;
    }
}

// ---------------- WMMA GEMM: out[b,256,NS] = W_eff[256,384] x s4[b,384,NS] + b_eff
// A slab (64x384 bf16, 48KB contiguous) loaded ONCE per block by the Tensor Data
// Mover with LDS padding (8 DWORDs per 64) to kill ds_load bank conflicts.
#define GNS 36864            /* 192*192 */
#define AROW 432             /* padded A row stride in elements: 192 DW -> 216 DW */
#define BROW 40              /* padded B row stride in elements */
__global__ void gemm_kernel(const unsigned short* __restrict__ Wbf,
                            const float* __restrict__ s4,
                            const float* __restrict__ beff,
                            float* __restrict__ out) {
    __shared__ __align__(128) unsigned short ldsA[64 * AROW];   // 55296 B
    __shared__ __align__(128) unsigned short ldsB[128 * BROW];  // 10240 B
    const int tid = threadIdx.x;
    const int wave = tid >> 5, lane = tid & 31;
    const int hi = lane >> 4, lm = lane & 15;
    const int wm = wave >> 2, wn = wave & 3;   // 2x4 wave grid -> 64x128 block tile
    const int n0 = blockIdx.x * 128;
    const int m0 = blockIdx.y * 64;
    const int b  = blockIdx.z;
    const float* Bsrc = s4 + (size_t)b * CCH * GNS;
    float* Obase = out + (size_t)b * OCH * GNS;

    // TDM: one 1D tile of 24576 bf16 elements (rows m0..m0+63 of W_eff are
    // contiguous), data_size=2B, pad_enable, interval=64 DW (code 5), amount=8 DW
    // (code 7). D# per CDNA5 ISA.
    if (wave == 0) {
        unsigned long long ga = (unsigned long long)(const void*)(Wbf + (size_t)m0 * CCH);
        unsigned lds0 = (unsigned)(unsigned long long)(const void*)&ldsA[0];
        u32x4 g0 = { 1u,                                   // count=1, user mode
                     lds0,                                 // lds_addr
                     (unsigned)ga,                         // global_addr[31:0]
                     ((unsigned)(ga >> 32) & 0x1FFFFFFu) | 0x80000000u }; // addr[56:32] | type=2
        u32x8 g1 = { (1u << 16) | (1u << 20) | (5u << 22) | (7u << 25), // data_size=2B, pad 8DW/64DW
                     0x60000000u,                          // tensor_dim0 = 24576 (lo16 in [31:16])
                     0x00010000u,                          // tensor_dim0 hi16=0, tensor_dim1 = 1
                     0x60000000u,                          // tile_dim0 = 24576
                     0u,                                   // tile_dim1 = 0 (unused), tile_dim2 = 0
                     24576u,                               // tensor_dim0_stride lo32
                     0u, 0u };
        asm volatile("tensor_load_to_lds %0, %1" :: "s"(g0), "s"(g1) : "memory");
    }

    f32x8 zero = {0.f, 0.f, 0.f, 0.f, 0.f, 0.f, 0.f, 0.f};
    f32x8 acc[2][2];
    acc[0][0] = zero; acc[0][1] = zero; acc[1][0] = zero; acc[1][1] = zero;

    // B-staging assignment: nn = tid&127 (coalesced), 8 k-pairs per thread
    const int nn  = tid & 127;
    const int kpb = (tid >> 7) * 8;

    for (int k0 = 0; k0 < CCH; k0 += 32) {
        // stage B tile: f32 -> bf16 pair-packed b32 stores, k-contiguous per column
        {
            const float* bp = Bsrc + (size_t)(k0 + 2 * kpb) * GNS + n0 + nn;
            unsigned* brow = (unsigned*)&ldsB[nn * BROW];
            #pragma unroll
            for (int q = 0; q < 8; ++q) {
                float f0 = bp[0];
                float f1 = bp[GNS];
                bp += 2 * (size_t)GNS;
                brow[kpb + q] = (unsigned)f2bf(f0) | ((unsigned)f2bf(f1) << 16);
            }
        }
        if (k0 == 0 && wave == 0) __builtin_amdgcn_s_wait_tensorcnt(0);
        __syncthreads();
        // fragments: lane holds 16 contiguous K (k = 16*hi + s; A and B use the same order).
        // A element offset accounts for the TDM pad: +16 elems per 128 elems.
        int ka = k0 + hi * 16;
        int kofs = ka + ((ka >> 7) << 4);
        v16bf a0 = *(const v16bf*)&ldsA[(wm * 32 + lm) * AROW + kofs];
        v16bf a1 = *(const v16bf*)&ldsA[(wm * 32 + 16 + lm) * AROW + kofs];
        v16bf b0 = *(const v16bf*)&ldsB[(wn * 32 + lm) * BROW + hi * 16];
        v16bf b1 = *(const v16bf*)&ldsB[(wn * 32 + 16 + lm) * BROW + hi * 16];
        acc[0][0] = __builtin_amdgcn_wmma_f32_16x16x32_bf16(false, a0, false, b0, (short)0, acc[0][0], false, false);
        acc[0][1] = __builtin_amdgcn_wmma_f32_16x16x32_bf16(false, a0, false, b1, (short)0, acc[0][1], false, false);
        acc[1][0] = __builtin_amdgcn_wmma_f32_16x16x32_bf16(false, a1, false, b0, (short)0, acc[1][0], false, false);
        acc[1][1] = __builtin_amdgcn_wmma_f32_16x16x32_bf16(false, a1, false, b1, (short)0, acc[1][1], false, false);
        __syncthreads();
    }
    // epilogue: C/D layout — VGPR r: lanes 0-15 M=r, lanes 16-31 M=r+8; N = lane%16
    #pragma unroll
    for (int mi = 0; mi < 2; ++mi) {
        #pragma unroll
        for (int ni = 0; ni < 2; ++ni) {
            #pragma unroll
            for (int r = 0; r < 8; ++r) {
                int m = m0 + wm * 32 + mi * 16 + r + hi * 8;
                int n = n0 + wn * 32 + ni * 16 + lm;
                Obase[(size_t)m * GNS + n] = acc[mi][ni][r] + beff[m];
            }
        }
    }
}

extern "C" void kernel_launch(void* const* d_in, const int* in_sizes, int n_in,
                              void* d_out, int out_size, void* d_ws, size_t ws_size,
                              hipStream_t stream) {
    (void)in_sizes; (void)n_in; (void)out_size; (void)ws_size;
    const float* source   = (const float*)d_in[0];   // [8,384,48,48]
    const float* guidance = (const float*)d_in[1];   // [8,3,768,768]
    const float* u1w1 = (const float*)d_in[2];
    const float* u1b1 = (const float*)d_in[3];
    const float* u1w2 = (const float*)d_in[4];
    const float* u1b2 = (const float*)d_in[5];
    const float* u1t  = (const float*)d_in[6];
    const float* u1s  = (const float*)d_in[7];
    const float* u2w1 = (const float*)d_in[8];
    const float* u2b1 = (const float*)d_in[9];
    const float* u2w2 = (const float*)d_in[10];
    const float* u2b2 = (const float*)d_in[11];
    const float* u2t  = (const float*)d_in[12];
    const float* u2s  = (const float*)d_in[13];
    const float* fixw = (const float*)d_in[14];
    const float* fixb = (const float*)d_in[15];
    const float* prjw = (const float*)d_in[16];
    const float* prjb = (const float*)d_in[17];
    float* outp = (float*)d_out;

    const int B = 8, C = 384;
    const int S1 = 96 * 96, S2 = 192 * 192;

    size_t off = 0;
    auto alloc = [&](size_t bytes) -> char* {
        char* p = (char*)d_ws + off;
        off += (bytes + 255) & ~(size_t)255;
        return p;
    };
    float* g1    = (float*)alloc((size_t)B * 3 * S1 * 4);
    float* g2    = (float*)alloc((size_t)B * 3 * S2 * 4);
    float* proj1 = (float*)alloc((size_t)B * KEYD * S1 * 4);
    float* proj2 = (float*)alloc((size_t)B * KEYD * S2 * 4);
    float* comb1 = (float*)alloc((size_t)B * 49 * S1 * 4);
    float* comb2 = (float*)alloc((size_t)B * 49 * S2 * 4);
    float* hr1   = (float*)alloc((size_t)B * C * S1 * 4);
    float* s2    = (float*)alloc((size_t)B * C * S1 * 4);
    float* hr2   = (float*)alloc((size_t)B * C * S2 * 4);
    float* s4    = (float*)alloc((size_t)B * C * S2 * 4);
    unsigned short* Wbf = (unsigned short*)alloc((size_t)OCH * CCH * 2);
    float* beff  = (float*)alloc((size_t)OCH * 4);

    const int T = 256;
    auto nb = [](int n) { return (n + 255) / 256; };

    // 1) pool guidance -> 96 and 192
    { int tot = B * 3 * S1; pool_kernel<<<nb(tot), T, 0, stream>>>(guidance, g1, 96, 96, 8, tot); }
    { int tot = B * 3 * S2; pool_kernel<<<nb(tot), T, 0, stream>>>(guidance, g2, 192, 192, 4, tot); }

    // ---- stage 1 (48 -> 96)
    { int tot = B * S1; rangeproj_kernel<<<nb(tot), T, 0, stream>>>(g1, u1w1, u1b1, u1w2, u1b2, proj1, S1, tot); }
    { int tot = B * S1; combined_kernel<<<nb(tot), T, 0, stream>>>(proj1, u1t, u1s, comb1, 96, 96, tot); }
    { int tot = B * C * S1; bicubic_kernel<<<nb(tot), T, 0, stream>>>(source, hr1, 48, 48, tot); }
    { int tot = B * S1; adaptive_kernel<<<nb(tot), T, 0, stream>>>(hr1, comb1, s2, C, 96, 96, tot); }

    // ---- stage 2 (96 -> 192)
    { int tot = B * S2; rangeproj_kernel<<<nb(tot), T, 0, stream>>>(g2, u2w1, u2b1, u2w2, u2b2, proj2, S2, tot); }
    { int tot = B * S2; combined_kernel<<<nb(tot), T, 0, stream>>>(proj2, u2t, u2s, comb2, 192, 192, tot); }
    { int tot = B * C * S2; bicubic_kernel<<<nb(tot), T, 0, stream>>>(s2, hr2, 96, 96, tot); }
    { int tot = B * S2; adaptive_kernel<<<nb(tot), T, 0, stream>>>(hr2, comb2, s4, C, 192, 192, tot); }

    // ---- fused fixup+proj: W_eff precompute then single WMMA GEMM
    { int tot = OCH * CCH; weff_kernel<<<nb(tot), T, 0, stream>>>(fixw, fixb, prjw, prjb, Wbf, beff); }
    {
        dim3 grid(S2 / 128, OCH / 64, B);   // (288, 4, 8)
        gemm_kernel<<<grid, T, 0, stream>>>(Wbf, s4, beff, outp);
    }
}
